// GIN_5970004541989
// MI455X (gfx1250) — matrix operand order repaired and checked by reading, not compiled, not run on hardware
//
#include <hip/hip_runtime.h>

// ---------------------------------------------------------------------------
// GIN forward on MI455X (gfx1250, wave32, WMMA).
//
// Pipeline (all on `stream`):
//   prep:   transpose+cvt weights fp32[K,N] -> fp16[N,K] (WMMA B layout);
//           build CSR of in-edges once (deg -> scan -> fill), reused by both convs.
//   conv1:  acc[i] = x[i] + sum_{j in N(i)} x[j]      (gather, no atomics)
//           h1 = relu(cvt16(acc) @ w1a + b1a)          [WMMA f16 -> f32 acc]
//           h2 = relu(h1 @ w1b + b1b)        (f32 out) [WMMA]
//   conv2:  acc[i] = h2[i] + sum h2[j]                 (gather)
//           h3 = relu(cvt16(acc) @ w2a + b2a)          [WMMA]
//           h4 = relu(h3 @ w2b + b2b)        (f16 out) [WMMA]
//   head:   per-node 256x10 dot -> atomic pool by graph; mean + log_softmax.
//
// GEMM tiling: block = 256 thr = 8 waves computes 64(M) x 256(N). A tile is
// staged into LDS as f16 once per block (row pad +8 halves vs bank aliasing),
// wave w owns columns 32w..32w+31 and runs 8 WMMAs per 32-wide k-step from
// ds_load_b128 fragments. M padded to 100032 = 1563*64 in workspace so every
// block is full (WMMA needs EXEC == all ones).
// ---------------------------------------------------------------------------

#define N_NODES   100000
#define NPAD      100032          // 1563 * 64
#define N_EDGES   1600000
#define D_FEAT    128
#define HIDDEN    256
#define N_CLASSES 10
#define N_GRAPHS  64

typedef __attribute__((ext_vector_type(16))) _Float16 v16h;
typedef __attribute__((ext_vector_type(8)))  _Float16 v8h;
typedef __attribute__((ext_vector_type(8)))  float    v8f;

// ---- helpers ---------------------------------------------------------------

__device__ __forceinline__ v16h frag_combine(v8h lo, v8h hi) {
    v16h r;
#pragma unroll
    for (int i = 0; i < 8; ++i) { r[i] = lo[i]; r[i + 8] = hi[i]; }
    return r;
}

// B fragment (32x16 f16): lane holds col lane%16, 16 consecutive K values.
__device__ __forceinline__ v16h load_b_frag(const _Float16* p) {
    return frag_combine(*(const v8h*)(p), *(const v8h*)(p + 8));
}

__device__ __forceinline__ v8h cvt8(float4 a, float4 b) {
    v8h h;
    h[0] = (_Float16)a.x; h[1] = (_Float16)a.y; h[2] = (_Float16)a.z; h[3] = (_Float16)a.w;
    h[4] = (_Float16)b.x; h[5] = (_Float16)b.y; h[6] = (_Float16)b.z; h[7] = (_Float16)b.w;
    return h;
}

__device__ __forceinline__ v8f wmma_f16(v16h a, v16h b, v8f c) {
    return __builtin_amdgcn_wmma_f32_16x16x32_f16(false, a, false, b, (short)0, c,
                                                  false, false);
}

// ---- GEMM: out[M,256] = relu(A[M,K] @ W[K,256] + bias) ---------------------
template <int K, bool A_F32, bool OUT_F32>
__global__ __launch_bounds__(256) void gemm_mlp_kernel(
    const void* __restrict__ Av, const _Float16* __restrict__ WT,
    const float* __restrict__ bias, void* __restrict__ Out)
{
    static_assert(K % 32 == 0, "K must be multiple of 32");
    constexpr int LDK = K + 8;                 // halves per LDS row (bank spread)
    __shared__ _Float16 As[64 * LDK];

    const int tid  = threadIdx.x;
    const int wave = tid >> 5;
    const int lane = tid & 31;
    const int l16  = lane & 15;
    const int half = lane >> 4;                // 0 or 1
    const int mb   = blockIdx.x * 64;
    const int n0   = wave * 32;

    // ---- stage A tile (64 x K) into LDS as f16, cooperatively ----
    constexpr int SEGS = 64 * K / 8;           // 8-half segments; SEGS % 256 == 0
#pragma unroll
    for (int sidx = tid; sidx < SEGS; sidx += 256) {
        const int row = sidx / (K / 8);
        const int seg = sidx - row * (K / 8);
        _Float16* dp = As + row * LDK + seg * 8;
        if constexpr (A_F32) {
            const float* sp = (const float*)Av + (size_t)(mb + row) * K + seg * 8;
            *(v8h*)dp = cvt8(*(const float4*)sp, *(const float4*)(sp + 4));
        } else {
            *(v8h*)dp = *(const v8h*)((const _Float16*)Av + (size_t)(mb + row) * K + seg * 8);
        }
    }
    __syncthreads();

    v8f c[4][2];
#pragma unroll
    for (int mi = 0; mi < 4; ++mi) { c[mi][0] = (v8f){}; c[mi][1] = (v8f){}; }

    const _Float16* b0p = WT + (size_t)(n0 + l16) * K + half * 16;
    const _Float16* b1p = WT + (size_t)(n0 + 16 + l16) * K + half * 16;
    const _Float16* a0  = As + l16 * LDK + half * 16;

#pragma unroll
    for (int k = 0; k < K; k += 32) {
        const v16h b0 = load_b_frag(b0p + k);
        const v16h b1 = load_b_frag(b1p + k);
#pragma unroll
        for (int mi = 0; mi < 4; ++mi) {
            const _Float16* ap = a0 + mi * 16 * LDK + k;
            const v16h a = frag_combine(*(const v8h*)ap, *(const v8h*)(ap + 8));
            c[mi][0] = wmma_f16(a, b0, c[mi][0]);
            c[mi][1] = wmma_f16(a, b1, c[mi][1]);
        }
    }

    const float bb0 = bias[n0 + l16];
    const float bb1 = bias[n0 + 16 + l16];

#pragma unroll
    for (int mi = 0; mi < 4; ++mi) {
#pragma unroll
        for (int v = 0; v < 8; ++v) {
            const int row = mb + mi * 16 + v + half * 8;   // C: VGPR v -> M=v (+8 hi)
            const float x0 = fmaxf(c[mi][0][v] + bb0, 0.0f);
            const float x1 = fmaxf(c[mi][1][v] + bb1, 0.0f);
            if constexpr (OUT_F32) {
                float* o = (float*)Out + (size_t)row * HIDDEN;
                o[n0 + l16]      = x0;
                o[n0 + 16 + l16] = x1;
            } else {
                _Float16* o = (_Float16*)Out + (size_t)row * HIDDEN;
                o[n0 + l16]      = (_Float16)x0;
                o[n0 + 16 + l16] = (_Float16)x1;
            }
        }
    }
}

// ---- CSR build -------------------------------------------------------------

__global__ __launch_bounds__(256) void count_kernel(const int* __restrict__ dst,
                                                    int* __restrict__ deg)
{
    const int e = blockIdx.x * blockDim.x + threadIdx.x;
    if (e < N_EDGES) atomicAdd(&deg[dst[e]], 1);
}

// inclusive scan per 256-block + per-block totals
__global__ __launch_bounds__(256) void scan1_kernel(const int* __restrict__ deg,
                                                    int* __restrict__ incl,
                                                    int* __restrict__ bsum)
{
    __shared__ int s[256];
    const int tid = threadIdx.x;
    const int i = blockIdx.x * 256 + tid;
    s[tid] = (i < N_NODES) ? deg[i] : 0;
    __syncthreads();
#pragma unroll
    for (int off = 1; off < 256; off <<= 1) {
        const int t = (tid >= off) ? s[tid - off] : 0;
        __syncthreads();
        s[tid] += t;
        __syncthreads();
    }
    if (i < N_NODES) incl[i] = s[tid];
    if (tid == 255) bsum[blockIdx.x] = s[255];
}

// exclusive scan of block sums (nb <= 512), single block
__global__ void scan2_kernel(int* __restrict__ bsum, int nb)
{
    __shared__ int s[512];
    const int tid = threadIdx.x;
    const int v = (tid < nb) ? bsum[tid] : 0;
    s[tid] = v;
    __syncthreads();
#pragma unroll
    for (int off = 1; off < 512; off <<= 1) {
        const int t = (tid >= off) ? s[tid - off] : 0;
        __syncthreads();
        s[tid] += t;
        __syncthreads();
    }
    if (tid < nb) bsum[tid] = s[tid] - v;   // exclusive
}

__global__ __launch_bounds__(256) void scan3_kernel(
    const int* __restrict__ deg, const int* __restrict__ incl,
    const int* __restrict__ bsum, int* __restrict__ rowstart,
    int* __restrict__ cursor)
{
    const int i = blockIdx.x * 256 + threadIdx.x;
    if (i < N_NODES) {
        const int rs = incl[i] - deg[i] + bsum[blockIdx.x];
        rowstart[i] = rs;
        cursor[i]   = rs;
    }
}

__global__ __launch_bounds__(256) void fill_kernel(
    const int* __restrict__ src, const int* __restrict__ dst,
    int* __restrict__ cursor, int* __restrict__ col)
{
    const int e = blockIdx.x * blockDim.x + threadIdx.x;
    if (e < N_EDGES) {
        const int p = atomicAdd(&cursor[dst[e]], 1);
        col[p] = src[e];
    }
}

// ---- gather aggregation: out[i] = feat[i] + sum_{j in N(i)} feat[j] --------
// One wave per node (D=128) or two (D=256); lane covers a float4 chunk.
template <int D>
__global__ __launch_bounds__(256) void agg_gather_kernel(
    const float* __restrict__ feat, const int* __restrict__ rowstart,
    const int* __restrict__ deg, const int* __restrict__ col,
    float* __restrict__ out)
{
    constexpr int WPN = D / 128;               // waves per node
    const int wid  = (blockIdx.x * 256 + threadIdx.x) >> 5;
    const int lane = threadIdx.x & 31;
    const int node = wid / WPN;
    const int part = wid - node * WPN;
    if (node >= N_NODES) return;               // wave-uniform
    const int cb = part * 128 + lane * 4;      // float column

    float4 acc = *(const float4*)(feat + (size_t)node * D + cb);   // self term
    const int s0 = rowstart[node];
    const int dn = deg[node];

    int j = 0;
    for (; j + 4 <= dn; j += 4) {
        const int n0 = col[s0 + j + 0];
        const int n1 = col[s0 + j + 1];
        const int n2 = col[s0 + j + 2];
        const int n3 = col[s0 + j + 3];
        const float4 v0 = *(const float4*)(feat + (size_t)n0 * D + cb);
        const float4 v1 = *(const float4*)(feat + (size_t)n1 * D + cb);
        const float4 v2 = *(const float4*)(feat + (size_t)n2 * D + cb);
        const float4 v3 = *(const float4*)(feat + (size_t)n3 * D + cb);
        acc.x += v0.x + v1.x + v2.x + v3.x;
        acc.y += v0.y + v1.y + v2.y + v3.y;
        acc.z += v0.z + v1.z + v2.z + v3.z;
        acc.w += v0.w + v1.w + v2.w + v3.w;
    }
    for (; j < dn; ++j) {
        const int n = col[s0 + j];
        const float4 v = *(const float4*)(feat + (size_t)n * D + cb);
        acc.x += v.x; acc.y += v.y; acc.z += v.z; acc.w += v.w;
    }
    *(float4*)(out + (size_t)node * D + cb) = acc;
}

// ---- weight prep: fp32 W[K,N] -> fp16 WT[N,K] ------------------------------
__global__ __launch_bounds__(256) void prep_wt_kernel(
    const float* __restrict__ w, _Float16* __restrict__ wt, int K, int N)
{
    const int idx = blockIdx.x * blockDim.x + threadIdx.x;
    if (idx >= K * N) return;
    const int k = idx / N;
    const int n = idx - k * N;
    wt[(size_t)n * K + k] = (_Float16)w[idx];
}

// ---- classifier + global mean pool (sum part) ------------------------------
__global__ __launch_bounds__(256) void classify_pool_kernel(
    const _Float16* __restrict__ h, const float* __restrict__ wlin,
    const float* __restrict__ blin, const int* __restrict__ batch,
    float* __restrict__ pooled, float* __restrict__ counts)
{
    const int node = blockIdx.x * blockDim.x + threadIdx.x;
    if (node >= N_NODES) return;
    float acc[N_CLASSES];
#pragma unroll
    for (int c = 0; c < N_CLASSES; ++c) acc[c] = 0.0f;

    const _Float16* hr = h + (size_t)node * HIDDEN;
    for (int k0 = 0; k0 < HIDDEN; k0 += 8) {
        const v8h hv = *(const v8h*)(hr + k0);
#pragma unroll
        for (int j = 0; j < 8; ++j) {
            const float x = (float)hv[j];
            const float* wrow = wlin + (size_t)(k0 + j) * N_CLASSES;  // lane-uniform
#pragma unroll
            for (int c = 0; c < N_CLASSES; ++c) acc[c] = fmaf(x, wrow[c], acc[c]);
        }
    }
    const int g = batch[node];
#pragma unroll
    for (int c = 0; c < N_CLASSES; ++c)
        unsafeAtomicAdd(&pooled[g * N_CLASSES + c], acc[c] + blin[c]);
    unsafeAtomicAdd(&counts[g], 1.0f);
}

// ---- mean + log_softmax ----------------------------------------------------
__global__ void finalize_kernel(const float* __restrict__ pooled,
                                const float* __restrict__ counts,
                                float* __restrict__ out)
{
    const int g = threadIdx.x;
    if (g >= N_GRAPHS) return;
    const float inv = 1.0f / fmaxf(counts[g], 1.0f);
    float v[N_CLASSES];
    float m = -3.402823466e38f;
#pragma unroll
    for (int c = 0; c < N_CLASSES; ++c) {
        v[c] = pooled[g * N_CLASSES + c] * inv;
        m = fmaxf(m, v[c]);
    }
    float s = 0.0f;
#pragma unroll
    for (int c = 0; c < N_CLASSES; ++c) s += __expf(v[c] - m);
    const float ls = __logf(s);
#pragma unroll
    for (int c = 0; c < N_CLASSES; ++c) out[g * N_CLASSES + c] = v[c] - m - ls;
}

// ---------------------------------------------------------------------------

extern "C" void kernel_launch(void* const* d_in, const int* in_sizes, int n_in,
                              void* d_out, int out_size, void* d_ws, size_t ws_size,
                              hipStream_t stream)
{
    const float* x     = (const float*)d_in[0];
    const int*   ei    = (const int*)d_in[1];
    const int*   src   = ei;
    const int*   dst   = ei + N_EDGES;
    const int*   batch = (const int*)d_in[2];
    const float* w1a = (const float*)d_in[3];
    const float* b1a = (const float*)d_in[4];
    const float* w1b = (const float*)d_in[5];
    const float* b1b = (const float*)d_in[6];
    const float* w2a = (const float*)d_in[7];
    const float* b2a = (const float*)d_in[8];
    const float* w2b = (const float*)d_in[9];
    const float* b2b = (const float*)d_in[10];
    const float* wlin = (const float*)d_in[11];
    const float* blin = (const float*)d_in[12];

    // ---- workspace carve-up (NPAD rows so every 64-row GEMM block is full)
    char* p = (char*)d_ws;
    float* acc = (float*)p;        p += (size_t)NPAD * HIDDEN * sizeof(float);     // 102.4 MB
    float* h2  = (float*)p;        p += (size_t)NPAD * HIDDEN * sizeof(float);     // 102.4 MB
    _Float16* hh = (_Float16*)p;   p += (size_t)NPAD * HIDDEN * sizeof(_Float16);  // 51.2 MB
    _Float16* wt1a = (_Float16*)p; p += (size_t)D_FEAT * HIDDEN * 2;
    _Float16* wt1b = (_Float16*)p; p += (size_t)HIDDEN * HIDDEN * 2;
    _Float16* wt2a = (_Float16*)p; p += (size_t)HIDDEN * HIDDEN * 2;
    _Float16* wt2b = (_Float16*)p; p += (size_t)HIDDEN * HIDDEN * 2;
    float* pooled = (float*)p;     p += (size_t)N_GRAPHS * N_CLASSES * sizeof(float);
    float* counts = (float*)p;     p += (size_t)N_GRAPHS * sizeof(float);
    int* deg      = (int*)p;       p += (size_t)N_NODES * sizeof(int);
    int* incl     = (int*)p;       p += (size_t)N_NODES * sizeof(int);
    int* rowstart = (int*)p;       p += (size_t)N_NODES * sizeof(int);
    int* cursor   = (int*)p;       p += (size_t)N_NODES * sizeof(int);
    int* bsum     = (int*)p;       p += (size_t)512 * sizeof(int);
    int* col      = (int*)p;       p += (size_t)N_EDGES * sizeof(int);
    _Float16* h4 = (_Float16*)acc; // acc dead once h3 exists -> reuse for h4

    const int gemm_grid = NPAD / 64;           // 1563
    const int nblk      = (N_NODES + 255) / 256;  // 391 (scan grid)
    const int eblk      = (N_EDGES + 255) / 256;

    // ---- prep: weights + CSR build + zero pooled ----
    prep_wt_kernel<<<(D_FEAT * HIDDEN + 255) / 256, 256, 0, stream>>>(w1a, wt1a, D_FEAT, HIDDEN);
    prep_wt_kernel<<<(HIDDEN * HIDDEN + 255) / 256, 256, 0, stream>>>(w1b, wt1b, HIDDEN, HIDDEN);
    prep_wt_kernel<<<(HIDDEN * HIDDEN + 255) / 256, 256, 0, stream>>>(w2a, wt2a, HIDDEN, HIDDEN);
    prep_wt_kernel<<<(HIDDEN * HIDDEN + 255) / 256, 256, 0, stream>>>(w2b, wt2b, HIDDEN, HIDDEN);
    hipMemsetAsync(pooled, 0, (size_t)(N_GRAPHS * N_CLASSES + N_GRAPHS) * sizeof(float), stream);
    hipMemsetAsync(deg, 0, (size_t)N_NODES * sizeof(int), stream);

    count_kernel<<<eblk, 256, 0, stream>>>(dst, deg);
    scan1_kernel<<<nblk, 256, 0, stream>>>(deg, incl, bsum);
    scan2_kernel<<<1, 512, 0, stream>>>(bsum, nblk);
    scan3_kernel<<<nblk, 256, 0, stream>>>(deg, incl, bsum, rowstart, cursor);
    fill_kernel<<<eblk, 256, 0, stream>>>(src, dst, cursor, col);

    // ---- conv1 ----
    agg_gather_kernel<D_FEAT><<<N_NODES / 8, 256, 0, stream>>>(x, rowstart, deg, col, acc);
    gemm_mlp_kernel<D_FEAT, true,  false><<<gemm_grid, 256, 0, stream>>>(acc, wt1a, b1a, hh);
    gemm_mlp_kernel<HIDDEN, false, true ><<<gemm_grid, 256, 0, stream>>>(hh,  wt1b, b1b, h2);

    // ---- conv2 ----
    agg_gather_kernel<HIDDEN><<<N_NODES / 4, 256, 0, stream>>>(h2, rowstart, deg, col, acc);
    gemm_mlp_kernel<HIDDEN, true,  false><<<gemm_grid, 256, 0, stream>>>(acc, wt2a, b2a, hh);
    gemm_mlp_kernel<HIDDEN, false, false><<<gemm_grid, 256, 0, stream>>>(hh,  wt2b, b2b, h4);

    // ---- head ----
    classify_pool_kernel<<<(N_NODES + 255) / 256, 256, 0, stream>>>(
        h4, wlin, blin, batch, pooled, counts);
    finalize_kernel<<<1, 64, 0, stream>>>(pooled, counts, (float*)d_out);
}
